// Encoder_45071386804587
// MI455X (gfx1250) — matrix operand order repaired and checked by reading, not compiled, not run on hardware
//
#include <hip/hip_runtime.h>
#include <hip/hip_bf16.h>

// ---------------------------------------------------------------------------
// 3-layer GRU encoder for MI455X (gfx1250, wave32, WMMA f16 16x16x32).
//
//   - prep: feats[T][B][128] = enc @ conv_w + conv_b (f16, time-major);
//           weights transposed+cast to f16 [N][K] (WMMA B layout)
//   - recurrent: 4 persistent WGs (batch is embarrassingly parallel in the
//     recurrence -> zero inter-WGP sync), 512 thr = 16 waves; wave w owns
//     gate column tile w*16 of z, r, hh. 2 workgroup barriers/layer/step.
//   - anti-LICM via an opaque integer offset (keeps global provenance) so
//     weight B-tiles stream from L2 each step instead of spilling to scratch;
//     B loads go through an explicit addrspace(1) cast -> global_load_b128
//     (LOADcnt only; no flat/DScnt contention with the LDS A-tile pipe).
//   - LDS h/rh rows padded to 272 halves (544 B): A-tile ds_load_b128s hit
//     distinct bank groups (2-way worst case instead of 16-way).
// ---------------------------------------------------------------------------

typedef __attribute__((ext_vector_type(16))) _Float16 v16h;
typedef __attribute__((ext_vector_type(8)))  _Float16 v8h;
typedef __attribute__((ext_vector_type(8)))  float    v8f;

#define H256   256
#define TSTEPS 1024
#define BATCH  64
#define HSTR   272   // padded LDS row stride (halves): 544 B

typedef const __attribute__((address_space(1))) v16h* gv16h_p;

__device__ __forceinline__ v8f wmma_f16(v16h a, v16h b, v8f c) {
  return __builtin_amdgcn_wmma_f32_16x16x32_f16(
      /*neg_a=*/false, a, /*neg_b=*/false, b,
      /*c_mod=*/(short)0, c, /*reuse_a=*/false, /*reuse_b=*/false);
}

// B-operand load, forced to the global path (global_load_b128 pair).
__device__ __forceinline__ v16h load_b_g(const _Float16* p) {
  return *(gv16h_p)(unsigned long long)(size_t)p;
}

// A-operand (16x32 f16): lane L: row = L%16; v[0..7] = K(kbase + (L/16)*16 ..),
// v[8..15] = K(kbase+8 + (L/16)*16 ..). Pairs of b128 coalesce into
// consecutive VGPRs (verified in round-1/2 asm: no shuffle movs).
__device__ __forceinline__ v16h load_a(const _Float16* p0) {
  v8h lo = *(const v8h*)(p0);
  v8h hi = *(const v8h*)(p0 + 8);
  v16h a;
#pragma unroll
  for (int i = 0; i < 8; i++) { a[i] = lo[i]; a[i + 8] = hi[i]; }
  return a;
}

__device__ __forceinline__ float sigm_f(float x) {
  return 1.0f / (1.0f + __expf(-x));       // x->-inf : 0, x->+inf : 1 (no NaN)
}
__device__ __forceinline__ float tanh_f(float x) {
  float e = __expf(2.0f * x);
  return 1.0f - 2.0f / (e + 1.0f);         // e->inf : 1, e->0 : -1 (no NaN)
}

// One GRU layer step for this wave's column tile.
template <int KX>
__device__ __forceinline__ void gru_layer(
    const _Float16* __restrict__ x, int xstride,   // rows 0..15, f16
    const _Float16* __restrict__ WT,               // [768][KX] f16 (global)
    const _Float16* __restrict__ UT,               // [768][256] f16 (global)
    float bz, float br, float bh,
    _Float16* h, _Float16* rh,                     // LDS [16][HSTR] f16
    int nbase, int lane,
    float* outp, int outstride,                    // nullable
    float* hfin)                                   // nullable, stride 256
{
  const int n      = lane & 15;
  const int kg     = lane >> 4;
  const int colabs = nbase + n;

  const _Float16* xa = x  + n * xstride + kg * 16;
  const _Float16* ha = h  + n * HSTR    + kg * 16;
  const _Float16* wz = WT + ((size_t)(      colabs)) * KX + kg * 16;
  const _Float16* wr = WT + ((size_t)(256 + colabs)) * KX + kg * 16;
  const _Float16* wh = WT + ((size_t)(512 + colabs)) * KX + kg * 16;
  const _Float16* uz = UT + ((size_t)(      colabs)) * 256 + kg * 16;
  const _Float16* ur = UT + ((size_t)(256 + colabs)) * 256 + kg * 16;
  const _Float16* uh = UT + ((size_t)(512 + colabs)) * 256 + kg * 16;

  // ---- merged: m = x@W (3 gates) and z/r recurrent = h@U, one k-loop so up
  // to 7 loads are in flight before the 5 WMMAs of each k-step ----
  v8f az = {}, ar = {}, ah = {}, vz = {}, vr = {};
#pragma unroll
  for (int k = 0; k < 8; k++) {
    v16h a_h = load_a(ha + k * 32);
    v16h buz = load_b_g(uz + k * 32);
    v16h bur = load_b_g(ur + k * 32);
    if (k < KX / 32) {
      v16h a_x = load_a(xa + k * 32);
      v16h bwz = load_b_g(wz + k * 32);
      v16h bwr = load_b_g(wr + k * 32);
      v16h bwh = load_b_g(wh + k * 32);
      az = wmma_f16(a_x, bwz, az);
      ar = wmma_f16(a_x, bwr, ar);
      ah = wmma_f16(a_x, bwh, ah);
    }
    vz = wmma_f16(a_h, buz, vz);
    vr = wmma_f16(a_h, bur, vr);
  }

  // ---- gates: C/D layout lane L vgpr v -> row = v + 8*(L/16), col = colabs ----
  float hv[8], zvv[8];
#pragma unroll
  for (int v = 0; v < 8; v++) {
    int row = v + 8 * kg;
    float hval = (float)h[row * HSTR + colabs];
    hv[v] = hval;
    float z = sigm_f(az[v] + vz[v] + bz);
    float r = sigm_f(ar[v] + vr[v] + br);
    zvv[v] = z;
    rh[row * HSTR + colabs] = (_Float16)(r * hval);
  }
  __syncthreads();   // all r*h written (and all h_old reads complete)

  // ---- candidate: (r*h) @ U[:, 2H:] ----
  v8f vh = {};
  const _Float16* ra = rh + n * HSTR + kg * 16;
#pragma unroll
  for (int k = 0; k < 8; k++) {
    v16h a = load_a(ra + k * 32);
    vh = wmma_f16(a, load_b_g(uh + k * 32), vh);
  }

  // ---- blend + state update ----
#pragma unroll
  for (int v = 0; v < 8; v++) {
    int row = v + 8 * kg;
    float cand = tanh_f(ah[v] + vh[v] + bh);
    float hn = zvv[v] * hv[v] + (1.0f - zvv[v]) * cand;
    h[row * HSTR + colabs] = (_Float16)hn;
    if (outp) outp[(size_t)row * outstride + colabs] = hn;
    if (hfin) hfin[row * 256 + colabs] = hn;
  }
  __syncthreads();   // h_new visible to next layer / next timestep
}

// ---------------------------------------------------------------------------
// Persistent 3-layer GRU: grid = 4 WGs (batch slices of 16), 512 thr = 16 waves
// ---------------------------------------------------------------------------
__global__ __launch_bounds__(512, 1) void gru3_kernel(
    const _Float16* __restrict__ feats,                    // [T][B][128] f16
    const _Float16* __restrict__ W0t, const _Float16* __restrict__ U0t,
    const _Float16* __restrict__ W1t, const _Float16* __restrict__ U1t,
    const _Float16* __restrict__ W2t, const _Float16* __restrict__ U2t,
    const float* __restrict__ b0, const float* __restrict__ b1,
    const float* __restrict__ b2, float* __restrict__ out)
{
  __shared__ _Float16 h0s[16 * HSTR];
  __shared__ _Float16 h1s[16 * HSTR];
  __shared__ _Float16 h2s[16 * HSTR];
  __shared__ _Float16 rhs[16 * HSTR];

  const int tid = threadIdx.x;
  for (int i = tid; i < 16 * HSTR; i += 512) {
    h0s[i] = (_Float16)0.0f; h1s[i] = (_Float16)0.0f; h2s[i] = (_Float16)0.0f;
  }
  __syncthreads();

  const int lane  = tid & 31;
  const int wave  = tid >> 5;
  const int nbase = wave * 16;
  const int bbase = blockIdx.x * 16;
  const int col   = nbase + (lane & 15);

  const float bz0 = b0[col], br0 = b0[256 + col], bh0 = b0[512 + col];
  const float bz1 = b1[col], br1 = b1[256 + col], bh1 = b1[512 + col];
  const float bz2 = b2[col], br2 = b2[256 + col], bh2 = b2[512 + col];

  const size_t OUTS = (size_t)BATCH * TSTEPS * H256;
  float* h0f = out + OUTS + (size_t)bbase * 256;
  float* h1f = h0f + (size_t)BATCH * 256;
  float* h2f = h1f + (size_t)BATCH * 256;

  // Opaque zero offset, refreshed every iteration: defeats LICM of the
  // weight B-tile loads (which would spill ~104 tiles to scratch) while
  // keeping the pointer chain rooted at the kernarg (global provenance).
  int lic = 0;

  for (int t = 0; t < TSTEPS; t++) {
    asm volatile("" : "+v"(lic));
    const _Float16* W0p = W0t + lic; const _Float16* U0p = U0t + lic;
    const _Float16* W1p = W1t + lic; const _Float16* U1p = U1t + lic;
    const _Float16* W2p = W2t + lic; const _Float16* U2p = U2t + lic;

    const _Float16* x0 = feats + ((size_t)t * BATCH + bbase) * 128;
    // Prefetch next timestep's feature slab (8192 halves; 512 thr * 16).
    if (t + 1 < TSTEPS)
      __builtin_prefetch(x0 + (size_t)BATCH * 128 + tid * 16, 0, 1);

    const bool last = (t == TSTEPS - 1);
    float* outp = out + ((size_t)bbase * TSTEPS + t) * H256;

    gru_layer<128>(x0, 128, W0p, U0p, bz0, br0, bh0, h0s, rhs, nbase, lane,
                   nullptr, 0, last ? h0f : nullptr);
    gru_layer<256>(h0s, HSTR, W1p, U1p, bz1, br1, bh1, h1s, rhs, nbase, lane,
                   nullptr, 0, last ? h1f : nullptr);
    gru_layer<256>(h1s, HSTR, W2p, U2p, bz2, br2, bh2, h2s, rhs, nbase, lane,
                   outp, TSTEPS * H256, last ? h2f : nullptr);
  }
}

// ---------------------------------------------------------------------------
// Prep: feats[t][b][c] = enc[b][t][:] @ conv_w + conv_b   (f32 math, f16 out)
// ---------------------------------------------------------------------------
__global__ void feats_kernel(const float* __restrict__ enc,
                             const float* __restrict__ cw,
                             const float* __restrict__ cb,
                             _Float16* __restrict__ feats)
{
  int idx = blockIdx.x * blockDim.x + threadIdx.x;   // T*B*128 = 8,388,608
  int c = idx & 127;
  int b = (idx >> 7) & 63;
  int t = idx >> 13;
  const float* e = enc + ((size_t)b * TSTEPS + t) * 40;
  float acc = cb[c];
#pragma unroll
  for (int d = 0; d < 40; d++) acc += e[d] * cw[d * 128 + c];
  feats[idx] = (_Float16)acc;
}

// Transpose+cast: in [K][768] f32 -> out [768][K] f16 (WMMA B-friendly)
__global__ void transpose_cast_kernel(const float* __restrict__ in,
                                      _Float16* __restrict__ out, int K)
{
  int idx = blockIdx.x * blockDim.x + threadIdx.x;   // 768*K
  int ncol = idx / K;
  int k    = idx % K;
  out[idx] = (_Float16)in[k * 768 + ncol];
}

// ---------------------------------------------------------------------------
extern "C" void kernel_launch(void* const* d_in, const int* in_sizes, int n_in,
                              void* d_out, int out_size, void* d_ws, size_t ws_size,
                              hipStream_t stream) {
  const float* enc = (const float*)d_in[0];
  const float* cw  = (const float*)d_in[1];
  const float* cb  = (const float*)d_in[2];
  const float* W0  = (const float*)d_in[3];
  const float* U0  = (const float*)d_in[4];
  const float* b0  = (const float*)d_in[5];
  const float* W1  = (const float*)d_in[6];
  const float* U1  = (const float*)d_in[7];
  const float* b1  = (const float*)d_in[8];
  const float* W2  = (const float*)d_in[9];
  const float* U2  = (const float*)d_in[10];
  const float* b2  = (const float*)d_in[11];
  float* out = (float*)d_out;

  // Scratch layout (f16 elements): ~18.9 MB total
  _Float16* ws    = (_Float16*)d_ws;
  _Float16* feats = ws;                                  // 1024*64*128
  _Float16* W0t   = feats + (size_t)TSTEPS * BATCH * 128;// 768*128
  _Float16* U0t   = W0t + 768 * 128;                     // 768*256
  _Float16* W1t   = U0t + 768 * 256;
  _Float16* U1t   = W1t + 768 * 256;
  _Float16* W2t   = U1t + 768 * 256;
  _Float16* U2t   = W2t + 768 * 256;

  feats_kernel<<<(TSTEPS * BATCH * 128) / 256, 256, 0, stream>>>(enc, cw, cb, feats);
  transpose_cast_kernel<<<(768 * 128) / 256, 256, 0, stream>>>(W0, W0t, 128);
  transpose_cast_kernel<<<(768 * 256) / 256, 256, 0, stream>>>(U0, U0t, 256);
  transpose_cast_kernel<<<(768 * 256) / 256, 256, 0, stream>>>(W1, W1t, 256);
  transpose_cast_kernel<<<(768 * 256) / 256, 256, 0, stream>>>(U1, U1t, 256);
  transpose_cast_kernel<<<(768 * 256) / 256, 256, 0, stream>>>(W2, W2t, 256);
  transpose_cast_kernel<<<(768 * 256) / 256, 256, 0, stream>>>(U2, U2t, 256);

  gru3_kernel<<<4, 512, 0, stream>>>(feats, W0t, U0t, W1t, U1t, W2t, U2t,
                                     b0, b1, b2, out);
}
